// ColorROUND_9861244911787
// MI455X (gfx1250) — compile-verified, matrix operand order checked by reading, not run on hardware
//
#include <hip/hip_runtime.h>

// Problem constants (from reference)
#define Bq 32
#define Sq 2048
#define Dq 8
#define Hq 256
#define NQ 256
#define Kq (7 * Hq)              // 1792
#define WSTEP 0.03125f
#define TWO_PI 6.28318530717958647692f
#define INV_TWO_PI 0.15915494309189533577f

typedef __attribute__((ext_vector_type(16))) __bf16          v16bf;
typedef __attribute__((ext_vector_type(16))) unsigned short  v16us;
typedef __attribute__((ext_vector_type(8)))  float           v8f;

__device__ __forceinline__ unsigned short f2bf_bits(float f) {
  union { float f; unsigned u; } v; v.f = f;
  unsigned u = v.u;
  unsigned r = u + 0x7FFFu + ((u >> 16) & 1u);   // round-to-nearest-even
  return (unsigned short)(r >> 16);
}

// gfx1250 async copy: global -> LDS, 16 bytes per lane, tracked by ASYNCcnt.
__device__ __forceinline__ void async_copy_b128(unsigned lds_off, unsigned goff,
                                                const void* sbase) {
  asm volatile("global_load_async_to_lds_b128 %0, %1, %2"
               :: "v"(lds_off), "v"(goff), "s"(sbase)
               : "memory");
}

// ---------------------------------------------------------------------------
// Phase 0: pre-convert Wr (f32) -> bf16 once (896 KB, stays L2-resident).
// ---------------------------------------------------------------------------
__global__ void __launch_bounds__(256)
k_cvt(const float* __restrict__ Wr, unsigned short* __restrict__ WrBf, int n) {
  int i = blockIdx.x * 256 + threadIdx.x;
  if (i < n) WrBf[i] = f2bf_bits(Wr[i]);
}

// ---------------------------------------------------------------------------
// Phase 1: delta[b,t,h] = wrap(x[b,t]·We[h] + be[h]);  is_rep[b,t]
// ---------------------------------------------------------------------------
__global__ void __launch_bounds__(256)
k_delta(const float* __restrict__ x, const float* __restrict__ We,
        const float* __restrict__ be, float* __restrict__ delta,
        float* __restrict__ rep) {
  const int bt = blockIdx.x;          // 0 .. B*S-1
  const int h  = threadIdx.x;         // 0 .. 255
  const int t  = bt & (Sq - 1);
  const float* xr = x + (size_t)bt * Dq;
  const float* wr = We + (size_t)h * Dq;
  float pt = be[h];
#pragma unroll
  for (int j = 0; j < Dq; ++j) pt += xr[j] * wr[j];
  // atan2(sin(pt), cos(pt)) == wrap pt into (-pi, pi]
  float d = pt - TWO_PI * rintf(pt * INV_TWO_PI);
  delta[(size_t)bt * Hq + h] = d;
  if (h == 0) {
    float r = 0.f;
    if (t != 0) {
      const float* xp = xr - Dq;
      bool same = true;
#pragma unroll
      for (int j = 0; j < Dq; ++j) same = same && (xr[j] == xp[j]);
      r = same ? 1.f : 0.f;
    }
    rep[bt] = r;
  }
}

// ---------------------------------------------------------------------------
// Phase 2: sequential scan over t, one thread per (b,h).
// ph += delta - sin(wb + STEP);  wb = wb + STEP + is_rep*delta
// ---------------------------------------------------------------------------
__global__ void __launch_bounds__(256)
k_scan(const float* __restrict__ delta, const float* __restrict__ rep,
       float* __restrict__ ph_hist, float* __restrict__ wb_hist) {
  const int b = blockIdx.x;           // 0 .. 31
  const int h = threadIdx.x;          // 0 .. 255
  float ph = 0.f, wb = 0.f;
  const size_t base = (size_t)b * Sq * Hq + h;
  const float* repb = rep + (size_t)b * Sq;
  for (int t = 0; t < Sq; ++t) {
    size_t idx = base + (size_t)t * Hq;
    float d   = delta[idx];
    float r   = repb[t];
    float wb2 = wb + WSTEP;
    ph = ph + d - __sinf(wb2);        // COUPLING = -1
    wb = wb2 + r * d;
    ph_hist[idx] = ph;
    wb_hist[idx] = wb;
  }
}

// ---------------------------------------------------------------------------
// Phase 3: logits = readout @ Wr^T + br, fused readout generation.
// M = 65536, N = 256, K = 1792.  Block: 256 threads = 8 waves,
// block tile 128(M) x 256(N).  Double-buffered async B staging, bf16 WMMA.
// ---------------------------------------------------------------------------
#define BPAD 40   // LDS row stride in bf16 (16B-aligned rows, conflict-friendly)

__global__ void __launch_bounds__(256)
k_gemm(const float* __restrict__ ph_hist, const float* __restrict__ wb_hist,
       const unsigned short* __restrict__ WrBf, const float* __restrict__ br,
       float* __restrict__ logits) {
  __shared__ unsigned short Bs[2][NQ][BPAD];     // 2 x 20 KB

  const int tid  = threadIdx.x;
  const int lane = tid & 31;
  const int wave = tid >> 5;
  const int half = lane >> 4;                    // k-offset selector (0 / 8)
  const int l15  = lane & 15;
  const int kb   = half * 8;
  const int rowM = blockIdx.x * 128 + wave * 16 + l15;   // A-matrix row per lane

  v8f acc[16];
#pragma unroll
  for (int i = 0; i < 16; ++i)
    acc[i] = (v8f){0.f, 0.f, 0.f, 0.f, 0.f, 0.f, 0.f, 0.f};

  // async-stage one 256x32 bf16 B slice (row per thread, 4 x b128 per thread)
  const unsigned ldsRow = (unsigned)(size_t)&Bs[0][tid][0];
  const unsigned ldsBufStride = (unsigned)(sizeof(unsigned short) * NQ * BPAD);
#define STAGE_B(bufi, K0s)                                                    \
  do {                                                                        \
    unsigned lds = ldsRow + (unsigned)(bufi) * ldsBufStride;                  \
    unsigned gof = (unsigned)(((size_t)tid * Kq + (size_t)(K0s)) * 2);        \
    async_copy_b128(lds +  0, gof +  0, WrBf);                                \
    async_copy_b128(lds + 16, gof + 16, WrBf);                                \
    async_copy_b128(lds + 32, gof + 32, WrBf);                                \
    async_copy_b128(lds + 48, gof + 48, WrBf);                                \
  } while (0)

  STAGE_B(0, 0);

  for (int ks = 0; ks < Kq / 32; ++ks) {
    const int K0  = ks * 32;
    const int cur = ks & 1;

    if (ks + 1 < Kq / 32) {
      STAGE_B(cur ^ 1, K0 + 32);                 // prefetch next slice
      asm volatile("s_wait_asynccnt 0x4" ::: "memory");  // current 4 landed
    } else {
      asm volatile("s_wait_asynccnt 0x0" ::: "memory");
    }
    __syncthreads();                             // all waves' slice visible

    // -- build A fragment (16-bit A 16x32 layout: runs {kb..kb+7},{kb+16..kb+23})
    const int seg   = K0 >> 8;                   // readout segment, uniform
    const int hbase = K0 & 255;
    const float* src = (seg == 4 || seg == 5) ? wb_hist : ph_hist;
    const float* p   = src + (size_t)rowM * Hq + hbase + kb;
    {
      // prefetch next K-step's A source while this slice computes
      const int K0n = K0 + 32;
      if (K0n < Kq) {
        const int segn = K0n >> 8;
        const float* srcn = (segn == 4 || segn == 5) ? wb_hist : ph_hist;
        __builtin_prefetch(srcn + (size_t)rowM * Hq + (K0n & 255) + kb, 0, 3);
      }
    }
    float av[16];
#pragma unroll
    for (int j = 0; j < 8; ++j) { av[j] = p[j]; av[8 + j] = p[16 + j]; }

    v16us abits;
    if (seg == 0 || seg == 4) {
#pragma unroll
      for (int j = 0; j < 16; ++j) abits[j] = f2bf_bits(__cosf(av[j]));
    } else if (seg == 1 || seg == 5) {
#pragma unroll
      for (int j = 0; j < 16; ++j) abits[j] = f2bf_bits(__sinf(av[j]));
    } else if (seg == 2) {
#pragma unroll
      for (int j = 0; j < 16; ++j) abits[j] = f2bf_bits(__cosf(0.5f * av[j]));
    } else if (seg == 3) {
#pragma unroll
      for (int j = 0; j < 16; ++j) abits[j] = f2bf_bits(__sinf(0.5f * av[j]));
    } else {                                     // seg 6: raw ph
#pragma unroll
      for (int j = 0; j < 16; ++j) abits[j] = f2bf_bits(av[j]);
    }
    const v16bf afrag = __builtin_bit_cast(v16bf, abits);

    // -- 16 WMMA tiles across N
#pragma unroll
    for (int nt = 0; nt < 16; ++nt) {
      const unsigned short* brow = &Bs[cur][nt * 16 + l15][kb];
      v16us bbits;
#pragma unroll
      for (int j = 0; j < 8; ++j) { bbits[j] = brow[j]; bbits[8 + j] = brow[16 + j]; }
      const v16bf bfrag = __builtin_bit_cast(v16bf, bbits);
      acc[nt] = __builtin_amdgcn_wmma_f32_16x16x32_bf16(
          false, afrag, false, bfrag, (short)0, acc[nt], false, false);
    }
    __syncthreads();                             // done reading before overwrite
  }

  // -- epilogue: C/D layout (lane = col, VGPR v = row v + 8*half) + bias
  const int rbase = blockIdx.x * 128 + wave * 16 + half * 8;
#pragma unroll
  for (int nt = 0; nt < 16; ++nt) {
    const int col  = nt * 16 + l15;
    const float bv = br[col];
#pragma unroll
    for (int v = 0; v < 8; ++v) {
      logits[(size_t)(rbase + v) * NQ + col] = acc[nt][v] + bv;
    }
  }
#undef STAGE_B
}

// ---------------------------------------------------------------------------
extern "C" void kernel_launch(void* const* d_in, const int* in_sizes, int n_in,
                              void* d_out, int out_size, void* d_ws, size_t ws_size,
                              hipStream_t stream) {
  const float* x  = (const float*)d_in[0];   // [B,S,D]
  const float* We = (const float*)d_in[1];   // [H,D]
  const float* be = (const float*)d_in[2];   // [H]
  const float* Wr = (const float*)d_in[3];   // [256, 7H]
  const float* br = (const float*)d_in[4];   // [256]

  float* out = (float*)d_out;
  const size_t n_logits = (size_t)Bq * Sq * NQ;    // 16,777,216
  const size_t n_hist   = (size_t)Bq * Sq * Hq;    // 16,777,216
  float* logits  = out;
  float* ph_hist = out + n_logits;
  float* wb_hist = out + n_logits + n_hist;

  // workspace: delta [B,S,H] f32 (64 MB) + is_rep [B,S] f32 (256 KB)
  //          + WrBf [256,1792] bf16 (896 KB)
  float* delta = (float*)d_ws;
  float* rep   = delta + (size_t)Bq * Sq * Hq;
  unsigned short* WrBf = (unsigned short*)(rep + (size_t)Bq * Sq);

  const int nWr = NQ * Kq;                          // 458752
  k_cvt  <<<(nWr + 255) / 256, 256, 0, stream>>>(Wr, WrBf, nWr);
  k_delta<<<Bq * Sq, 256, 0, stream>>>(x, We, be, delta, rep);
  k_scan <<<Bq, 256, 0, stream>>>(delta, rep, ph_hist, wb_hist);
  k_gemm <<<(Bq * Sq) / 128, 256, 0, stream>>>(ph_hist, wb_hist, WrBf, br, logits);
}
